// AttentionAggregator_85315230368142
// MI455X (gfx1250) — matrix operand order repaired and checked by reading, not compiled, not run on hardware
//
#include <hip/hip_runtime.h>
#include <stdint.h>

#define GN 16384
#define GM 4096
#define GD 128
#define RB 32          // rows per block
#define STEPS 128      // GM / 32
#define ASTRIDE 20     // u32 stride of one A-tile row in LDS (conflict-avoid pad)

typedef __bf16 v16bf __attribute__((ext_vector_type(16)));
typedef float  v8f   __attribute__((ext_vector_type(8)));
typedef unsigned int v8u __attribute__((ext_vector_type(8)));

__device__ __forceinline__ unsigned short f2bf(float f) {
  unsigned u = __builtin_bit_cast(unsigned, f);
  u += 0x7fffu + ((u >> 16) & 1u);          // round-to-nearest-even
  return (unsigned short)(u >> 16);
}

__device__ __forceinline__ v16bf mk16(uint4 a, uint4 b) {
  v8u u;
  u[0] = a.x; u[1] = a.y; u[2] = a.z; u[3] = a.w;
  u[4] = b.x; u[5] = b.y; u[6] = b.z; u[7] = b.w;
  return __builtin_bit_cast(v16bf, u);
}

// ---------------- Kernel 1: s_self = X@a[:D], s_neigh = V@a[D:] --------------
__global__ void __launch_bounds__(256)
gat_dots(const float* __restrict__ sf, const float* __restrict__ fn,
         const float* __restrict__ a, float* __restrict__ ss, float* __restrict__ sn) {
  int gw   = (blockIdx.x * 256 + threadIdx.x) >> 5;  // one wave per row
  int lane = threadIdx.x & 31;
  const float* frow;
  const float* av;
  float* o;
  if (gw < GN) { frow = sf + (size_t)gw * GD; av = a;      o = ss + gw; }
  else         { frow = fn + (size_t)(gw - GN) * GD; av = a + GD; o = sn + (gw - GN); }
  float4 f  = ((const float4*)frow)[lane];
  float4 aa = ((const float4*)av)[lane];
  float d = f.x * aa.x + f.y * aa.y + f.z * aa.z + f.w * aa.w;
#pragma unroll
  for (int off = 16; off; off >>= 1) d += __shfl_xor(d, off);
  if (lane == 0) *o = d;
}

// ---------------- Kernel 2: pack mask int32 -> bits (wave32 ballot) ----------
__global__ void __launch_bounds__(256)
gat_pack(const int* __restrict__ nm, unsigned* __restrict__ pm) {
  int wid  = (blockIdx.x * 256 + threadIdx.x) >> 5;
  int lane = threadIdx.x & 31;
  int nw   = (gridDim.x * 256) >> 5;
  const size_t words = (size_t)GN * (GM / 32);
  for (size_t W = wid; W < words; W += nw) {
    int v = nm[W * 32 + lane];
    unsigned long long b = __ballot(v != 0);
    if (lane == 0) pm[W] = (unsigned)b;
  }
}

// ---------------- Kernel 3: V(f32) -> bf16 B-fragment-ready layout -----------
// vpack[s][w][q][lane] = 8 bf16: K = 32s + ((lane&16)>>1) + 16q + t, col = 16w + (lane&15)
__global__ void __launch_bounds__(256)
gat_vpack(const float* __restrict__ fn, uint4* __restrict__ vp) {
  int idx  = blockIdx.x * 256 + threadIdx.x;   // 0..65535
  int lane = idx & 31;
  int q    = (idx >> 5) & 1;
  int w    = (idx >> 6) & 7;
  int s    = idx >> 9;
  int col  = (w << 4) + (lane & 15);
  int kb   = (s << 5) + ((lane & 16) >> 1) + (q << 4);
  unsigned short h[8];
#pragma unroll
  for (int t = 0; t < 8; ++t) h[t] = f2bf(fn[(size_t)(kb + t) * GD + col]);
  uint4 u;
  u.x = (unsigned)h[0] | ((unsigned)h[1] << 16);
  u.y = (unsigned)h[2] | ((unsigned)h[3] << 16);
  u.z = (unsigned)h[4] | ((unsigned)h[5] << 16);
  u.w = (unsigned)h[6] | ((unsigned)h[7] << 16);
  vp[idx] = u;
}

// ---------------- Kernel 4: fused masked-softmax @ V (flash-style) -----------
__global__ void __launch_bounds__(256)
gat_main(const float* __restrict__ ss, const float* __restrict__ sn,
         const unsigned* __restrict__ pm, const uint4* __restrict__ vp,
         float* __restrict__ out) {
  __shared__ float tlds[GM];                       // 16 KB
  __shared__ unsigned pmlds[RB * 129];             // 16.5 KB, pad for banks
  __shared__ float srow[RB];
  __shared__ float zlds[RB];
  __shared__ __align__(16) unsigned atile[2][RB * ASTRIDE];  // double-buffered bf16 A tile

  const int tid  = threadIdx.x;
  const int lane = tid & 31;
  const int wave = tid >> 5;
  const int r0   = blockIdx.x * RB;

  {  // prologue: stage t, packed mask rows, s rows
    const float4* t4 = (const float4*)sn;
    float4* l4 = (float4*)tlds;
#pragma unroll
    for (int k = 0; k < 4; ++k) l4[tid + 256 * k] = t4[tid + 256 * k];
#pragma unroll
    for (int k = 0; k < 16; ++k) {
      int idx = tid + 256 * k;  // 0..4095 over RB x 128 words
      pmlds[(idx >> 7) * 129 + (idx & 127)] = pm[(size_t)r0 * 128 + idx];
    }
    if (tid < RB) srow[tid] = ss[r0 + tid];
  }
  __syncthreads();

  // producer role: thread -> (row prow, 4 consecutive j's at offset jq in each step)
  const int prow = tid >> 3;
  const int jq   = (tid & 7) * 4;
  const float sr = srow[prow];
  const unsigned* pmrow = &pmlds[prow * 129];
  const float4* tq = (const float4*)&tlds[jq];
  float zp = 0.f;

  // consumer role: wave -> 16 output cols; lane -> A rows (m0, m0+16)
  const int m0  = lane & 15;
  const int kh4 = (lane >> 4) * 4;                 // u32 offset into A row
  const uint4* bbase = vp + (size_t)wave * 64 + lane;

  v8f c0 = {}; v8f c1 = {};

  for (int s = 0; s < STEPS; ++s) {
    // ---- produce 4 attn values (unnormalized exp) ----
    unsigned mw = pmrow[s] >> jq;
    float4 tv = tq[s * 8];
    float x0 = sr + tv.x, x1 = sr + tv.y, x2 = sr + tv.z, x3 = sr + tv.w;
    float p0 = __expf(fmaxf(x0, 0.2f * x0));
    float p1 = __expf(fmaxf(x1, 0.2f * x1));
    float p2 = __expf(fmaxf(x2, 0.2f * x2));
    float p3 = __expf(fmaxf(x3, 0.2f * x3));
    p0 = (mw & 1u) ? p0 : 0.f;
    p1 = (mw & 2u) ? p1 : 0.f;
    p2 = (mw & 4u) ? p2 : 0.f;
    p3 = (mw & 8u) ? p3 : 0.f;
    zp += (p0 + p1) + (p2 + p3);
    unsigned lo = (unsigned)f2bf(p0) | ((unsigned)f2bf(p1) << 16);
    unsigned hi = (unsigned)f2bf(p2) | ((unsigned)f2bf(p3) << 16);
    *(uint2*)&atile[s & 1][prow * ASTRIDE + (tid & 7) * 2] = make_uint2(lo, hi);
    __syncthreads();

    // ---- consume: A frags (2 row tiles) + B frag, 2 WMMAs ----
    const unsigned* ar0 = &atile[s & 1][m0 * ASTRIDE + kh4];
    uint4 a0lo = *(const uint4*)ar0;
    uint4 a0hi = *(const uint4*)(ar0 + 8);
    const unsigned* ar1 = ar0 + 16 * ASTRIDE;
    uint4 a1lo = *(const uint4*)ar1;
    uint4 a1hi = *(const uint4*)(ar1 + 8);
    uint4 b0 = bbase[(size_t)s * 512];
    uint4 b1 = bbase[(size_t)s * 512 + 32];
    v16bf A0 = mk16(a0lo, a0hi);
    v16bf A1 = mk16(a1lo, a1hi);
    v16bf B  = mk16(b0, b1);
    c0 = __builtin_amdgcn_wmma_f32_16x16x32_bf16(false, A0, false, B, (short)0, c0, false, false);
    c1 = __builtin_amdgcn_wmma_f32_16x16x32_bf16(false, A1, false, B, (short)0, c1, false, false);
  }

  // ---- reduce Z across the 8 producer lanes of each row ----
  zp += __shfl_xor(zp, 1);
  zp += __shfl_xor(zp, 2);
  zp += __shfl_xor(zp, 4);
  if ((tid & 7) == 0) zlds[prow] = zp;
  __syncthreads();

  // ---- normalize and store (C/D layout: VGPR e -> row e (+8 for hi lanes)) ----
  const int col = (wave << 4) + (lane & 15);
  const int hi8 = (lane >> 4) << 3;
#pragma unroll
  for (int e = 0; e < 8; ++e) {
    int lr0 = e + hi8;
    float z0 = zlds[lr0];
    out[(size_t)(r0 + lr0) * GD + col] = (z0 > 0.f) ? (c0[e] / z0) : 0.f;
    int lr1 = lr0 + 16;
    float z1 = zlds[lr1];
    out[(size_t)(r0 + lr1) * GD + col] = (z1 > 0.f) ? (c1[e] / z1) : 0.f;
  }
}

// ---------------- launch -----------------------------------------------------
extern "C" void kernel_launch(void* const* d_in, const int* in_sizes, int n_in,
                              void* d_out, int out_size, void* d_ws, size_t ws_size,
                              hipStream_t stream) {
  (void)in_sizes; (void)n_in; (void)out_size; (void)ws_size;
  const float* self_feats = (const float*)d_in[0];
  const float* fneigh     = (const float*)d_in[1];
  const int*   nmat       = (const int*)d_in[2];
  const float* avec       = (const float*)d_in[3];
  float* out = (float*)d_out;

  char* ws = (char*)d_ws;
  float*    ss = (float*)ws;                                   // N f32      (64 KB)
  float*    sn = (float*)(ws + 65536);                         // M f32      (16 KB)
  unsigned* pm = (unsigned*)(ws + 81920);                      // N*128 u32  (8 MB)
  uint4*    vp = (uint4*)(ws + 81920 + 8388608);               // 1 MB bf16 B-pack

  gat_dots <<<(GN + GM) / 8, 256, 0, stream>>>(self_feats, fneigh, avec, ss, sn);
  gat_pack <<<1024, 256, 0, stream>>>(nmat, pm);
  gat_vpack<<<256, 256, 0, stream>>>(fneigh, vp);
  gat_main <<<GN / RB, 256, 0, stream>>>(ss, sn, pm, vp, out);
}
